// TSPRLModel_13778255086207
// MI455X (gfx1250) — matrix operand order repaired and checked by reading, not compiled, not run on hardware
//
#include <hip/hip_runtime.h>

// Problem constants (from reference): B=32, P=256, N=1000, D=512
#define B_    32
#define P_    256
#define N_    1000
#define D_    512
#define NPAD  1024          // N padded to multiple of 16 for WMMA tiling
#define ALPHA 0.8f
#define BETA  0.2f

typedef __attribute__((ext_vector_type(16))) __bf16 v16bf;
typedef __attribute__((ext_vector_type(8)))  float  v8f;

union FragBF {              // one WMMA bf16 operand per lane: 16 bf16 = 32B = 8 VGPRs
    v16bf v;
    uint4 q[2];
};

union HalfRow {             // 8 contiguous bf16 = 16B vector store
    __bf16 h[8];
    uint4  q;
};

// ---------------------------------------------------------------------------
// Kernel 1: node_features[b, n, :] = points[b,n,:] @ W_emb + b_emb  (bf16)
// Rows n in [N_, NPAD) are zero-filled so GEMM B-fragment loads need no
// bounds checks. One thread produces 8 contiguous bf16 via one b128 store.
// ---------------------------------------------------------------------------
__global__ __launch_bounds__(256) void tsprl_embed_bf16(
    const float* __restrict__ pts,     // (B, N, 2)
    const float* __restrict__ W,       // (2, D)
    const float* __restrict__ bias,    // (D,)
    __bf16* __restrict__ nf)           // (B, NPAD, D) bf16
{
    int idx = blockIdx.x * blockDim.x + threadIdx.x;   // B*NPAD*(D/8) threads
    int dg  = (idx & 63) * 8;                          // D/8 = 64 groups
    int n   = (idx >> 6) & (NPAD - 1);
    int b   = idx >> 16;                               // 64 * 1024 = 2^16

    HalfRow row;
    if (n < N_) {
        float x0 = pts[((size_t)b * N_ + n) * 2 + 0];
        float x1 = pts[((size_t)b * N_ + n) * 2 + 1];
#pragma unroll
        for (int j = 0; j < 8; ++j) {
            int d = dg + j;
            row.h[j] = (__bf16)(x0 * W[d] + x1 * W[D_ + d] + bias[d]);
        }
    } else {
#pragma unroll
        for (int j = 0; j < 8; ++j) row.h[j] = (__bf16)0.0f;
    }
    *(uint4*)(nf + (((size_t)b * NPAD + n) * D_ + dg)) = row.q;
}

// ---------------------------------------------------------------------------
// Kernel 2: fused  sim = cf @ nf^T  (WMMA bf16 -> f32)  + gather/scale/mask.
//   grid  = B * (P/16) blocks (one 16-row M-tile of one batch per block)
//   block = 256 threads = 8 waves; each wave owns N-tiles wave, wave+8, ...
// A rows are gathered through current_node at fragment-load time; the
// epilogue is branch-free (clause-able gathers + v_cndmask select) with the
// scattered adj/vis cachelines prefetched before the K loop.
// ---------------------------------------------------------------------------
__global__ __launch_bounds__(256) void tsprl_sim_wmma(
    const __bf16* __restrict__ nf,     // (B, NPAD, D) bf16
    const float*  __restrict__ adj,    // (B, N, N)
    const int*    __restrict__ cur,    // (B, P)
    const int*    __restrict__ vis,    // (B, P, N)  0/1
    float*        __restrict__ out)    // (B, P, N)
{
    const int b     = blockIdx.x >> 4;       // 512 blocks = 32 batches * 16 tiles
    const int mtile = blockIdx.x & 15;
    const int lane  = threadIdx.x & 31;
    const int wave  = threadIdx.x >> 5;      // 0..7
    const int l15   = lane & 15;
    const int khalf = lane >> 4;             // 0: lanes 0-15, 1: lanes 16-31

    // A-matrix (16x32 bf16): lane holds row M = lane&15 in both halves,
    // K elems {khalf*8..+7} then {16+khalf*8..+7} (ISA 7.12.2). Row is the
    // gathered node_features row for pomo index p = mtile*16 + (lane&15).
    const int pA   = mtile * 16 + l15;
    const int rowA = cur[b * P_ + pA];       // current node id, 0..N-1
    const __bf16* aRow  = nf + ((size_t)b * NPAD + rowA) * D_;
    const int     aOff0 = khalf * 8;
    const int     aOff1 = 16 + khalf * 8;

    // Tile-invariant epilogue row offsets (all fit in int32: max ~33e6 elems).
    // Output row r of the D-tile is m = r + 8*khalf  ->  p = mtile*16 + m.
    int adjOff[8];   // element offset of adj[b, cur[b,p], 0]
    int pvOff[8];    // element offset of vis/out[b, p, 0]
#pragma unroll
    for (int r = 0; r < 8; ++r) {
        const int p  = mtile * 16 + khalf * 8 + r;
        const int cn = cur[b * P_ + p];
        adjOff[r] = (b * N_ + cn) * N_;
        pvOff[r]  = (b * P_ + p) * N_;
    }

    for (int nt = wave; nt < 63; nt += 8) {  // 63 tiles cover n in [0,1008)
        const int nBase = nt * 16;
        const int n     = nBase + l15;
        const int nc    = (n < N_) ? n : 0;  // clamp (no EXEC change near WMMA)

        // Pull the scattered epilogue cachelines toward the caches while the
        // K loop runs (global_prefetch_b8; no LOADcnt, no wait needed).
#pragma unroll
        for (int r = 0; r < 8; ++r) {
            __builtin_prefetch(adj + (size_t)(adjOff[r] + nc), 0, 3);
            __builtin_prefetch(vis + (size_t)(pvOff[r] + nc), 0, 3);
        }

        // B-matrix (32x16 bf16) = nf^T tile: lane holds col N = lane&15,
        // K elems {khalf*8..+7} then {16+khalf*8..+7} within one nf row.
        const __bf16* bRow =
            nf + ((size_t)b * NPAD + nBase + l15) * D_ + khalf * 8;

        v8f acc = {};
#pragma unroll 4
        for (int kk = 0; kk < D_; kk += 32) {
            FragBF fa, fb;
            fa.q[0] = *(const uint4*)(aRow + kk + aOff0);
            fa.q[1] = *(const uint4*)(aRow + kk + aOff1);
            fb.q[0] = *(const uint4*)(bRow + kk);
            fb.q[1] = *(const uint4*)(bRow + kk + 16);
            acc = __builtin_amdgcn_wmma_f32_16x16x32_bf16(
                false, fa.v, false, fb.v, (short)0, acc, false, false);
        }

        // Branch-free epilogue: gather everything first (one load clause),
        // then select + store. Lanes with n >= N_ skip via the outer if only.
        if (n < N_) {
            float av[8];
            int   mv[8];
#pragma unroll
            for (int r = 0; r < 8; ++r) {
                av[r] = adj[(size_t)(adjOff[r] + n)];
                mv[r] = vis[(size_t)(pvOff[r] + n)];
            }
#pragma unroll
            for (int r = 0; r < 8; ++r) {
                const float val = mv[r] ? -__builtin_inff()
                                        : fmaf(BETA, acc[r], ALPHA * av[r]);
                out[(size_t)(pvOff[r] + n)] = val;
            }
        }
    }
}

// ---------------------------------------------------------------------------
extern "C" void kernel_launch(void* const* d_in, const int* in_sizes, int n_in,
                              void* d_out, int out_size, void* d_ws, size_t ws_size,
                              hipStream_t stream) {
    const float* points = (const float*)d_in[0];   // (B, N, 2)
    const float* adj    = (const float*)d_in[1];   // (B, N, N)
    const int*   cur    = (const int*)d_in[2];     // (B, P)
    const int*   vis    = (const int*)d_in[3];     // (B, P, N)
    const float* W      = (const float*)d_in[4];   // (2, D)
    const float* bias   = (const float*)d_in[5];   // (D,)
    float*       out    = (float*)d_out;           // (B, P, N)

    // Workspace: bf16 node_features, (B, NPAD, D) = 32 MB (fully initialized).
    __bf16* nf = (__bf16*)d_ws;

    const int embedThreads = B_ * NPAD * (D_ / 8);          // 2,097,152
    tsprl_embed_bf16<<<embedThreads / 256, 256, 0, stream>>>(points, W, bias, nf);

    tsprl_sim_wmma<<<B_ * (P_ / 16), 256, 0, stream>>>(nf, adj, cur, vis, out);
}
